// SpikingCNN_429496730122
// MI455X (gfx1250) — matrix operand order repaired and checked by reading, not compiled, not run on hardware
//
#include <hip/hip_runtime.h>
#include <hip/hip_bf16.h>

typedef _Float16 v16h __attribute__((ext_vector_type(16)));
typedef _Float16 v8h  __attribute__((ext_vector_type(8)));
typedef float    v8f  __attribute__((ext_vector_type(8)));

#define T_STEPS 20
#define BETA_F  0.9f
#define THR_F   1.0f
#define BB      8192
#define DD      784
#define DP      800      // D padded to 25 chunks of K=32
#define DPS     808      // ws1 row stride (halves): 404 dwords -> bank-conflict-free
#define HH      128
#define HS      136      // ws2 / wspk row stride (halves): 68 dwords -> conflict-free
#define OO      10
#define WAVES   4        // waves per block (128 blocks x 4 waves = 512 row tiles)
#define ROWS_PER_WAVE 16

// cheap deterministic per-(row,col,t) hash -> 4 uniform bytes
__device__ __forceinline__ unsigned hash3(unsigned a, unsigned b, unsigned c) {
  unsigned h = a * 0x9E3779B1u + b * 0x85EBCA77u + c * 0xC2B2AE3Du;
  h ^= h >> 16; h *= 0x7FEB352Du; h ^= h >> 15;
  return h;
}

__global__ void __launch_bounds__(32 * WAVES)
snn_wmma_kernel(const float* __restrict__ x,
                const float* __restrict__ w1,
                const float* __restrict__ b1,
                const float* __restrict__ w2,
                const float* __restrict__ b2,
                float* __restrict__ out) {
  extern __shared__ _Float16 smem[];
  _Float16* ws1     = smem;                 // [128][DPS] f16 W1 (cols 784..799 zero)
  _Float16* ws2     = ws1 + HH * DPS;       // [16][HS]   f16 W2 (rows 10..15 zero)
  _Float16* wspkAll = ws2 + 16 * HS;        // [WAVES][16][HS] spk1 staging

  const int tid  = threadIdx.x;
  const int nthr = 32 * WAVES;

  // ---- stage W1 into LDS: fp32 -> f16, pad cols 784..799 with zero ----
  for (int g = tid; g < HH * (DP / 16); g += nthr) {
    int n  = g / (DP / 16);
    int kg = (g - n * (DP / 16)) * 16;
    _Float16* dst = ws1 + n * DPS + kg;
    if (kg < DD) {
      const float* src = w1 + n * DD + kg;
#pragma unroll
      for (int j = 0; j < 16; ++j) dst[j] = (_Float16)src[j];
    } else {
#pragma unroll
      for (int j = 0; j < 16; ++j) dst[j] = (_Float16)0.f;
    }
  }
  // ---- stage W2 into LDS, pad output rows 10..15 with zero ----
  for (int i = tid; i < 16 * HH; i += nthr) {
    int n = i >> 7, k = i & 127;
    ws2[n * HS + k] = (n < OO) ? (_Float16)w2[n * HH + k] : (_Float16)0.f;
  }
  __syncthreads();

  const int wave = tid >> 5;
  const int lane = tid & 31;
  const int half = lane >> 4;   // which 16-lane half of the wave
  const int lm   = lane & 15;
  _Float16* wspk = wspkAll + wave * 16 * HS;

  const int rowTile = blockIdx.x * WAVES + wave;   // 128 blocks * 4 waves = 512
  const int r0   = rowTile * ROWS_PER_WAVE;
  const int arow = r0 + lm;                        // row this lane's A elems use
  const float* xrow = x + (long)arow * DD;

  // per-lane biases for the 8 H-tiles (C-fragment column = nt*16 + lm)
  float bb1[8];
#pragma unroll
  for (int nt = 0; nt < 8; ++nt) bb1[nt] = b1[nt * 16 + lm];
  const float bb2 = (lm < OO) ? b2[lm] : 0.f;

  const v8f vzero = {0.f, 0.f, 0.f, 0.f, 0.f, 0.f, 0.f, 0.f};
  v8f mem1[8];
#pragma unroll
  for (int nt = 0; nt < 8; ++nt) mem1[nt] = vzero;
  v8f mem2 = vzero;
  v8f acc  = vzero;

  for (int t = 0; t < T_STEPS; ++t) {
    // ---------------- layer 1: cur1 = spikes @ W1^T ----------------
    v8f cur[8];
#pragma unroll
    for (int nt = 0; nt < 8; ++nt) cur[nt] = vzero;

    for (int kc = 0; kc < DP / 32; ++kc) {
      // A fragment (16x32 f16): half 0 -> K {0..7,16..23}, half 1 -> K {8..15,24..31}
      const int kb0 = kc * 32 + half * 8;
      const int kb1 = kb0 + 16;
      float xv[16];
#pragma unroll
      for (int j = 0; j < 8; ++j) xv[j] = xrow[kb0 + j];   // kb0 <= 776: always valid
      if (kb1 < DD) {
#pragma unroll
        for (int j = 0; j < 8; ++j) xv[8 + j] = xrow[kb1 + j];
      } else {
#pragma unroll
        for (int j = 0; j < 8; ++j) xv[8 + j] = -1.f;      // spike prob 0 in pad region
      }

      v16h a;
#pragma unroll
      for (int j = 0; j < 16; j += 4) {
        unsigned c0 = (unsigned)((j < 8) ? (kb0 + j) : (kb1 + (j - 8)));
        unsigned h  = hash3((unsigned)arow, c0, (unsigned)t);
#pragma unroll
        for (int q = 0; q < 4; ++q) {
          float u = (float)((h >> (8 * q)) & 0xFFu);
          a[j + q] = (u < xv[j + q] * 256.0f) ? (_Float16)1.f : (_Float16)0.f;
        }
      }

#pragma unroll
      for (int nt = 0; nt < 8; ++nt) {
        // B fragment (32x16 f16): half 0 -> K {0..7,16..23}, half 1 -> K {8..15,24..31}, N=lm
        const _Float16* bp = ws1 + (nt * 16 + lm) * DPS + kc * 32 + half * 8;
        v8h blo = *(const v8h*)bp;
        v8h bhi = *(const v8h*)(bp + 16);
        v16h bf = __builtin_shufflevector(blo, bhi,
                                          0, 1, 2, 3, 4, 5, 6, 7,
                                          8, 9, 10, 11, 12, 13, 14, 15);
        cur[nt] = __builtin_amdgcn_wmma_f32_16x16x32_f16(
            false, a, false, bf, (short)0, cur[nt], false, false);
      }
    }

    // ---- LIF layer 1 + stage spk1 (C layout -> LDS row-major) ----
#pragma unroll
    for (int nt = 0; nt < 8; ++nt) {
#pragma unroll
      for (int i = 0; i < 8; ++i) {
        float m = mem1[nt][i];
        float c = cur[nt][i] + bb1[nt];
        float reset = (m > THR_F) ? THR_F : 0.f;   // heaviside(m - thr) * thr
        m = BETA_F * m + c - reset;
        float s = (m > THR_F) ? 1.f : 0.f;
        mem1[nt][i] = m;
        wspk[(i + half * 8) * HS + nt * 16 + lm] = (_Float16)s;
      }
    }
    __syncthreads();

    // ---------------- layer 2: cur2 = spk1 @ W2^T ----------------
    v8f cur2 = vzero;
#pragma unroll
    for (int kc = 0; kc < HH / 32; ++kc) {
      const _Float16* ap = wspk + lm * HS + kc * 32 + half * 8;
      v8h alo = *(const v8h*)ap;
      v8h ahi = *(const v8h*)(ap + 16);
      v16h a2 = __builtin_shufflevector(alo, ahi,
                                        0, 1, 2, 3, 4, 5, 6, 7,
                                        8, 9, 10, 11, 12, 13, 14, 15);
      const _Float16* bp = ws2 + lm * HS + kc * 32 + half * 8;
      v8h blo = *(const v8h*)bp;
      v8h bhi = *(const v8h*)(bp + 16);
      v16h bf = __builtin_shufflevector(blo, bhi,
                                        0, 1, 2, 3, 4, 5, 6, 7,
                                        8, 9, 10, 11, 12, 13, 14, 15);
      cur2 = __builtin_amdgcn_wmma_f32_16x16x32_f16(
          false, a2, false, bf, (short)0, cur2, false, false);
    }

    // ---- LIF layer 2 + accumulate output spikes ----
#pragma unroll
    for (int i = 0; i < 8; ++i) {
      float m = mem2[i];
      float c = cur2[i] + bb2;
      float reset = (m > THR_F) ? THR_F : 0.f;
      m = BETA_F * m + c - reset;
      float s = (m > THR_F) ? 1.f : 0.f;
      mem2[i] = m;
      acc[i] += s;
    }
    __syncthreads();   // protect wspk WAR across steps
  }

  // ---- write out: [8192,10] = acc / NUM_STEPS ----
  if (lm < OO) {
#pragma unroll
    for (int i = 0; i < 8; ++i) {
      out[(long)(r0 + i + half * 8) * OO + lm] = acc[i] * (1.0f / T_STEPS);
    }
  }
}

extern "C" void kernel_launch(void* const* d_in, const int* in_sizes, int n_in,
                              void* d_out, int out_size, void* d_ws, size_t ws_size,
                              hipStream_t stream) {
  (void)in_sizes; (void)n_in; (void)d_ws; (void)ws_size; (void)out_size;
  const float* x  = (const float*)d_in[0];   // [8192,784]
  const float* w1 = (const float*)d_in[1];   // [128,784]
  const float* b1 = (const float*)d_in[2];   // [128]
  const float* w2 = (const float*)d_in[3];   // [10,128]
  const float* b2 = (const float*)d_in[4];   // [10]
  float* out = (float*)d_out;                // [8192,10]

  const size_t lds_bytes =
      (size_t)(HH * DPS + 16 * HS + WAVES * 16 * HS) * sizeof(_Float16); // ~229 KB < 320 KB
  dim3 grid(BB / (ROWS_PER_WAVE * WAVES));   // 128 blocks (1 WGP each, LDS-bound)
  dim3 block(32 * WAVES);                    // 128 threads = 4 waves
  snn_wmma_kernel<<<grid, block, lds_bytes, stream>>>(x, w1, b1, w2, b2, out);
}
